// CausalCreditAssigner_86320252715294
// MI455X (gfx1250) — matrix operand order repaired and checked by reading, not compiled, not run on hardware
//
#include <hip/hip_runtime.h>
#include <hip/hip_bf16.h>

// CausalCreditAssigner on MI455X (gfx1250): f16-WMMA GEMM pipeline, wave32.
// b128 fragment traffic, pre-swizzled weights, A-fragments cached in VGPRs,
// intervention-MLP chain fused through LDS.
#define B_  16
#define S_  1024
#define H_  256
#define NH_ 4
#define DH_ 64
#define BS_ (B_ * S_)

typedef __attribute__((ext_vector_type(16))) _Float16 v16h;
typedef __attribute__((ext_vector_type(8)))  float    v8f;

union Frag16 { v16h h; uint4 q[2]; };

static __device__ __forceinline__ v8f wmma_f16(v16h a, v16h b, v8f c) {
  // D = A(16x32 f16) * B(32x16 f16) + C(16x16 f32)
  return __builtin_amdgcn_wmma_f32_16x16x32_f16(false, a, false, b, (short)0, c,
                                                false, false);
}

// ---- CDNA5 WMMA fragment loaders (contiguous-k order, wave32) -------------
// A 16x32 f16 from row-major [M][K]: lane L holds row m=L&15; its 16 halves
// are contiguous at k = k0+16*(L>>4).
static __device__ __forceinline__ v16h load_a(const _Float16* __restrict__ p,
                                              int ld, int k0) {
  const int lane = threadIdx.x & 31;
  const int m  = lane & 15;
  const int hf = lane >> 4;
  const _Float16* row = p + (size_t)m * ld + k0 + 16 * hf;
  Frag16 f;
  f.q[0] = *(const uint4*)(row);
  f.q[1] = *(const uint4*)(row + 8);
  return f.h;
}

// B 32x16 f16 from N-major (row-major [N][K]) source (K^T for QK^T):
// lane holds col n=L&15; 16 contiguous halves at k0+16*(L>>4).
static __device__ __forceinline__ v16h load_bT(const _Float16* __restrict__ M,
                                               int ldk, int n0, int k0) {
  const int lane = threadIdx.x & 31;
  const int n  = lane & 15;
  const int hf = lane >> 4;
  const _Float16* row = M + (size_t)(n0 + n) * ldk + k0 + 16 * hf;
  Frag16 f;
  f.q[0] = *(const uint4*)(row);
  f.q[1] = *(const uint4*)(row + 8);
  return f.h;
}

// B 32x16 f16 from pre-swizzled weight: [kc][nt][lane][16 halves] contiguous.
static __device__ __forceinline__ v16h load_bs(const _Float16* __restrict__ Wsw,
                                               int NT, int kc, int nt) {
  const int lane = threadIdx.x & 31;
  const _Float16* p = Wsw + ((size_t)(kc * NT + nt) * 32 + lane) * 16;
  Frag16 f;
  f.q[0] = *(const uint4*)(p);
  f.q[1] = *(const uint4*)(p + 8);
  return f.h;
}

// Stage a contiguous 16x256 f16 tile (8 KB) into LDS with b128 loads.
static __device__ __forceinline__ void stage_tile(const _Float16* __restrict__ src,
                                                  _Float16* __restrict__ Xs) {
  const uint4* s = (const uint4*)src;
  uint4* d = (uint4*)Xs;
  d[threadIdx.x]       = s[threadIdx.x];
  d[threadIdx.x + 256] = s[threadIdx.x + 256];
}

// ---- conversion / init kernels --------------------------------------------
__global__ __launch_bounds__(256) void cvt4_f16_kernel(const float* __restrict__ src,
                                                       _Float16* __restrict__ dst,
                                                       int n4) {
  int i = blockIdx.x * 256 + threadIdx.x;
  if (i < n4) {
    float4 v = ((const float4*)src)[i];
    union { _Float16 h[4]; uint2 u; } r;
    r.h[0] = (_Float16)v.x; r.h[1] = (_Float16)v.y;
    r.h[2] = (_Float16)v.z; r.h[3] = (_Float16)v.w;
    ((uint2*)dst)[i] = r.u;
  }
}

// f32 weight [256][N] row-major -> f16 swizzled fragment order.
// dest((kc,nt,lane,e)) with lane = (n&15) + 16*((k&31)>>4), e = k&15.
__global__ __launch_bounds__(256) void swz_b_kernel(const float* __restrict__ W,
                                                    _Float16* __restrict__ Wsw,
                                                    int N, int NT, int nt_off) {
  int i = blockIdx.x * 256 + threadIdx.x;
  if (i >= 256 * N) return;
  int k = i / N, n = i % N;
  int kc = k >> 5, kk = k & 31;
  int lane = (n & 15) + 16 * (kk >> 4);
  size_t dst = ((size_t)(kc * NT + (n >> 4) + nt_off) * 32 + lane) * 16 + (kk & 15);
  Wsw[dst] = (_Float16)W[i];
}

__global__ __launch_bounds__(256) void concat_bias_kernel(const float* __restrict__ bq,
                                                          const float* __restrict__ bk,
                                                          float* __restrict__ bqk) {
  int i = blockIdx.x * 256 + threadIdx.x;       // 512
  bqk[i] = (i < 256) ? bq[i] : bk[i - 256];
}

__global__ __launch_bounds__(256) void zero_f32_kernel(float* __restrict__ p, int n) {
  int i = blockIdx.x * 256 + threadIdx.x;
  if (i < n) p[i] = 0.f;
}

// ---- fused Q|K projection: [16384,256] @ [256,512] + bias -> f16 ----------
__global__ __launch_bounds__(256) void proj_qk_kernel(
    const _Float16* __restrict__ Xh, const _Float16* __restrict__ Wqks,
    const float* __restrict__ bqk, _Float16* __restrict__ QKh) {
  __shared__ __align__(16) _Float16 Xs[16 * 256];
  const int tile = blockIdx.x;                 // 1024 tiles of 16 rows
  stage_tile(Xh + (size_t)tile * 16 * 256, Xs);
  __syncthreads();
  const int wave = threadIdx.x >> 5;
  const int lane = threadIdx.x & 31;
  const int nl = lane & 15, hf = lane >> 4;
  v16h af[8];                                  // cache A fragments in VGPRs
#pragma unroll
  for (int kc = 0; kc < 8; ++kc) af[kc] = load_a(Xs, 256, kc * 32);
#pragma unroll
  for (int t = 0; t < 4; ++t) {
    const int nt = wave * 4 + t;
    const int n0 = nt * 16;
    const float bias = bqk[n0 + nl];
    v8f acc;
#pragma unroll
    for (int r = 0; r < 8; ++r) acc[r] = bias;
#pragma unroll
    for (int kc = 0; kc < 8; ++kc)
      acc = wmma_f16(af[kc], load_bs(Wqks, 32, kc, nt), acc);
    _Float16* out = QKh + (size_t)tile * 16 * 512 + n0 + nl;
#pragma unroll
    for (int r = 0; r < 8; ++r)
      out[(size_t)(r + 8 * hf) * 512] = (_Float16)acc[r];
  }
}

// ---- attention: flash-style softmax column-sum accumulation ---------------
// block = (b, head, 16-row q tile); wave w owns key columns [w*128, w*128+128)
__global__ __launch_bounds__(256) void attn_colsum_kernel(
    const _Float16* __restrict__ QKh, float* __restrict__ aw_sum) {
  __shared__ float wmaxS[8 * 16], wsumS[8 * 16];
  __shared__ float rowmaxS[16], rinvS[16];
  const int blk = blockIdx.x;
  const int qt = blk & 63;
  const int h  = (blk >> 6) & (NH_ - 1);
  const int b  = blk >> 8;
  const int wave = threadIdx.x >> 5;
  const int lane = threadIdx.x & 31;
  const int nl = lane & 15, hf = lane >> 4;

  const _Float16* Qb = QKh + (size_t)(b * S_ + qt * 16) * 512 + h * DH_;
  const _Float16* Kb = QKh + (size_t)(b * S_) * 512 + 256 + h * DH_;

  const v16h a0 = load_a(Qb, 512, 0);
  const v16h a1 = load_a(Qb, 512, 32);

  float sc[8][8];                 // scores kept in VGPRs (64 regs/lane)
  float lmax[8], lsum[8];
#pragma unroll
  for (int r = 0; r < 8; ++r) { lmax[r] = -3.0e38f; lsum[r] = 0.f; }

#pragma unroll
  for (int jt = 0; jt < 8; ++jt) {
    const int j0 = wave * 128 + jt * 16;
    v8f acc = {0.f, 0.f, 0.f, 0.f, 0.f, 0.f, 0.f, 0.f};
    acc = wmma_f16(a0, load_bT(Kb, 512, j0, 0), acc);
    acc = wmma_f16(a1, load_bT(Kb, 512, j0, 32), acc);
#pragma unroll
    for (int r = 0; r < 8; ++r) {
      const float v = acc[r] * 0.125f;          // 1/sqrt(DH)
      sc[jt][r] = v;
      const float nm = fmaxf(lmax[r], v);
      lsum[r] = lsum[r] * __expf(lmax[r] - nm) + __expf(v - nm);
      lmax[r] = nm;
    }
  }
  // online-softmax reduce across the 16 column-lanes of each half
#pragma unroll
  for (int msk = 1; msk < 16; msk <<= 1) {
#pragma unroll
    for (int r = 0; r < 8; ++r) {
      const float om = __shfl_xor(lmax[r], msk, 32);
      const float os = __shfl_xor(lsum[r], msk, 32);
      const float nm = fmaxf(lmax[r], om);
      lsum[r] = lsum[r] * __expf(lmax[r] - nm) + os * __expf(om - nm);
      lmax[r] = nm;
    }
  }
  if (nl == 0) {
#pragma unroll
    for (int r = 0; r < 8; ++r) {
      wmaxS[wave * 16 + r + 8 * hf] = lmax[r];
      wsumS[wave * 16 + r + 8 * hf] = lsum[r];
    }
  }
  __syncthreads();
  if (threadIdx.x < 16) {                        // combine the 8 waves
    const int m = threadIdx.x;
    float fm = wmaxS[m], fs = wsumS[m];
    for (int w = 1; w < 8; ++w) {
      const float om = wmaxS[w * 16 + m], os = wsumS[w * 16 + m];
      const float nm = fmaxf(fm, om);
      fs = fs * __expf(fm - nm) + os * __expf(om - nm);
      fm = nm;
    }
    rowmaxS[m] = fm;
    rinvS[m] = 1.0f / fs;
  }
  __syncthreads();
  float rm[8], ri[8];
#pragma unroll
  for (int r = 0; r < 8; ++r) { rm[r] = rowmaxS[r + 8 * hf]; ri[r] = rinvS[r + 8 * hf]; }
#pragma unroll
  for (int jt = 0; jt < 8; ++jt) {
    float colp = 0.f;
#pragma unroll
    for (int r = 0; r < 8; ++r)
      colp += __expf(sc[jt][r] - rm[r]) * ri[r];
    atomicAdd(&aw_sum[b * S_ + wave * 128 + jt * 16 + nl], colp);
  }
}

// ---- strengths = sigmoid(relu(X@Wm1+bm1)@Wm2+bm2) -------------------------
__global__ __launch_bounds__(256) void strengths_kernel(
    const _Float16* __restrict__ Xh, const _Float16* __restrict__ Wm1s,
    const float* __restrict__ bm1, const float* __restrict__ Wm2,
    const float* __restrict__ bm2, float* __restrict__ strengths) {
  __shared__ __align__(16) _Float16 Xs[16 * 256];
  __shared__ float Hs[16 * 128];
  __shared__ float red[16 * 16];
  const int tile = blockIdx.x;
  stage_tile(Xh + (size_t)tile * 16 * 256, Xs);
  __syncthreads();
  const int wave = threadIdx.x >> 5;
  const int lane = threadIdx.x & 31;
  const int nl = lane & 15, hf = lane >> 4;
  const int n0 = wave * 16;
  v8f acc;
#pragma unroll
  for (int r = 0; r < 8; ++r) acc[r] = bm1[n0 + nl];
#pragma unroll
  for (int kc = 0; kc < 8; ++kc)
    acc = wmma_f16(load_a(Xs, 256, kc * 32), load_bs(Wm1s, 8, kc, wave), acc);
  const float w2 = Wm2[n0 + nl];
#pragma unroll
  for (int r = 0; r < 8; ++r)
    Hs[(r + 8 * hf) * 128 + n0 + nl] = fmaxf(acc[r], 0.f) * w2;
  __syncthreads();
  const int row = threadIdx.x >> 4, p = threadIdx.x & 15;
  float s = 0.f;
#pragma unroll
  for (int i = 0; i < 8; ++i) s += Hs[row * 128 + p * 8 + i];
  red[row * 16 + p] = s;
  __syncthreads();
  if (p == 0) {
    float t = bm2[0];
    for (int i = 0; i < 16; ++i) t += red[row * 16 + i];
    strengths[tile * 16 + row] = 1.0f / (1.0f + __expf(-t));
  }
}

// ---- colmean via suffix-sum; rescale aw_sum -> aw_mean in place -----------
__global__ __launch_bounds__(256) void colmean_kernel(
    float* __restrict__ aw, const float* __restrict__ strengths,
    float* __restrict__ colmean) {
  __shared__ float ss[S_];
  const int b = blockIdx.x;
  for (int j = threadIdx.x; j < S_; j += 256) ss[j] = strengths[b * S_ + j];
  __syncthreads();
  if (threadIdx.x == 0) {
    float run = 0.f;
    for (int j = S_ - 1; j >= 0; --j) { run += ss[j]; ss[j] = run; }
  }
  __syncthreads();
  const float sc = 1.0f / (float)(NH_ * S_);
  for (int j = threadIdx.x; j < S_; j += 256) {
    const float am = aw[b * S_ + j] * sc;
    aw[b * S_ + j] = am;                              // aw_mean
    colmean[b * S_ + j] = am * ss[j] * (1.0f / (float)S_);
  }
}

// ---- cm[b,i,j] = aw_mean[b,j]*strengths[b,i]*(j<=i): 64 MB streaming write
__global__ __launch_bounds__(256) void cm_kernel(
    const float* __restrict__ aw_mean, const float* __restrict__ strengths,
    float* __restrict__ cm) {
  const int bi = blockIdx.x;               // b*S + i
  const int b = bi >> 10, i = bi & 1023;
  const float st = strengths[bi];
  const float4* awv = (const float4*)(aw_mean + (size_t)b * S_);
  float4* out = (float4*)(cm + (size_t)bi * S_);
  const int c = threadIdx.x;               // 256 threads x float4 = 1024 floats
  const float4 a = awv[c];
  const int j = c * 4;
  float4 v;
  v.x = (j + 0 <= i) ? a.x * st : 0.f;
  v.y = (j + 1 <= i) ? a.y * st : 0.f;
  v.z = (j + 2 <= i) ? a.z * st : 0.f;
  v.w = (j + 3 <= i) ? a.w * st : 0.f;
  out[c] = v;
}

// ---- base[b,h] = (1/S) * sum_s x[b,s,h]*colmean[b,s] ----------------------
__global__ __launch_bounds__(256) void base_kernel(
    const float* __restrict__ x, const float* __restrict__ colmean,
    float* __restrict__ base) {
  const int b = blockIdx.x, h = threadIdx.x;
  const float* xb = x + (size_t)b * S_ * H_;
  float acc = 0.f;
  for (int s = 0; s < S_; ++s)
    acc += xb[(size_t)s * H_ + h] * colmean[b * S_ + s];
  base[b * H_ + h] = acc * (1.0f / (float)S_);
}

// ---- fused intervention MLP + credit head ---------------------------------
// H = relu(X@Wi1+bi1); F = base + 0.1*tanh(H@Wi2+bi2)*colmean/S;
// credit = mask ? fo - (relu(F@We1+be1)@We2+be2) : 0.
// All three GEMMs chained through LDS; X never leaves the block.
__global__ __launch_bounds__(256) void mlp_credit_kernel(
    const _Float16* __restrict__ Xh,
    const _Float16* __restrict__ Wi1s, const float* __restrict__ bi1,
    const _Float16* __restrict__ Wi2s, const float* __restrict__ bi2,
    const _Float16* __restrict__ We1s, const float* __restrict__ be1,
    const float* __restrict__ We2, const float* __restrict__ be2,
    const float* __restrict__ base, const float* __restrict__ colmean,
    const float* __restrict__ fo, const unsigned char* __restrict__ mask,
    float* __restrict__ credit) {
  __shared__ __align__(16) _Float16 Xs[16 * 256];
  __shared__ __align__(16) _Float16 Hs2[16 * 256];
  __shared__ __align__(16) _Float16 Fs[16 * 256];
  __shared__ float Ps[16 * 128];
  __shared__ float red[16 * 16];
  __shared__ float cmv[16];
  const int tile = blockIdx.x;
  const int b = tile >> 6;
  stage_tile(Xh + (size_t)tile * 16 * 256, Xs);
  if (threadIdx.x < 16)
    cmv[threadIdx.x] = colmean[tile * 16 + threadIdx.x] * (1.0f / (float)S_);
  __syncthreads();
  const int wave = threadIdx.x >> 5;
  const int lane = threadIdx.x & 31;
  const int nl = lane & 15, hf = lane >> 4;

  v16h af[8];
#pragma unroll
  for (int kc = 0; kc < 8; ++kc) af[kc] = load_a(Xs, 256, kc * 32);
  // ---- stage 1: hidden = relu(X@Wi1+bi1) -> LDS f16
#pragma unroll
  for (int t = 0; t < 2; ++t) {
    const int nt = wave * 2 + t;
    const int n0 = nt * 16;
    v8f acc;
#pragma unroll
    for (int r = 0; r < 8; ++r) acc[r] = bi1[n0 + nl];
#pragma unroll
    for (int kc = 0; kc < 8; ++kc)
      acc = wmma_f16(af[kc], load_bs(Wi1s, 16, kc, nt), acc);
#pragma unroll
    for (int r = 0; r < 8; ++r)
      Hs2[(r + 8 * hf) * 256 + n0 + nl] = (_Float16)fmaxf(acc[r], 0.f);
  }
  __syncthreads();
#pragma unroll
  for (int kc = 0; kc < 8; ++kc) af[kc] = load_a(Hs2, 256, kc * 32);
  // ---- stage 2: feats = base + 0.1*tanh(H@Wi2+bi2)*cmv -> LDS f16
#pragma unroll
  for (int t = 0; t < 2; ++t) {
    const int nt = wave * 2 + t;
    const int n0 = nt * 16;
    v8f acc;
#pragma unroll
    for (int r = 0; r < 8; ++r) acc[r] = bi2[n0 + nl];
#pragma unroll
    for (int kc = 0; kc < 8; ++kc)
      acc = wmma_f16(af[kc], load_bs(Wi2s, 16, kc, nt), acc);
    const float bs = base[b * H_ + n0 + nl];
#pragma unroll
    for (int r = 0; r < 8; ++r) {
      const float d = 0.1f * tanhf(acc[r]);
      Fs[(r + 8 * hf) * 256 + n0 + nl] = (_Float16)(bs + d * cmv[r + 8 * hf]);
    }
  }
  __syncthreads();
#pragma unroll
  for (int kc = 0; kc < 8; ++kc) af[kc] = load_a(Fs, 256, kc * 32);
  // ---- stage 3: relu(F@We1+be1) * We2 -> row reduce
  {
    const int n0 = wave * 16;
    v8f acc;
#pragma unroll
    for (int r = 0; r < 8; ++r) acc[r] = be1[n0 + nl];
#pragma unroll
    for (int kc = 0; kc < 8; ++kc)
      acc = wmma_f16(af[kc], load_bs(We1s, 8, kc, wave), acc);
    const float w2 = We2[n0 + nl];
#pragma unroll
    for (int r = 0; r < 8; ++r)
      Ps[(r + 8 * hf) * 128 + n0 + nl] = fmaxf(acc[r], 0.f) * w2;
  }
  __syncthreads();
  const int row = threadIdx.x >> 4, p = threadIdx.x & 15;
  float s = 0.f;
#pragma unroll
  for (int i = 0; i < 8; ++i) s += Ps[row * 128 + p * 8 + i];
  red[row * 16 + p] = s;
  __syncthreads();
  if (p == 0) {
    float o = be2[0];
    for (int i = 0; i < 16; ++i) o += red[row * 16 + i];
    const int gr = tile * 16 + row;
    credit[gr] = mask[gr] ? (fo[gr >> 10] - o) : 0.f;
  }
}

// ---------------------------------------------------------------------------
extern "C" void kernel_launch(void* const* d_in, const int* in_sizes, int n_in,
                              void* d_out, int out_size, void* d_ws, size_t ws_size,
                              hipStream_t stream) {
  const float* x    = (const float*)d_in[0];
  const float* fo   = (const float*)d_in[1];
  const unsigned char* mask = (const unsigned char*)d_in[2];  // jnp.bool_ = 1 byte
  const float* Wq  = (const float*)d_in[3];  const float* bq  = (const float*)d_in[4];
  const float* Wk  = (const float*)d_in[5];  const float* bk  = (const float*)d_in[6];
  const float* Wm1 = (const float*)d_in[7];  const float* bm1 = (const float*)d_in[8];
  const float* Wm2 = (const float*)d_in[9];  const float* bm2 = (const float*)d_in[10];
  const float* Wi1 = (const float*)d_in[11]; const float* bi1 = (const float*)d_in[12];
  const float* Wi2 = (const float*)d_in[13]; const float* bi2 = (const float*)d_in[14];
  const float* We1 = (const float*)d_in[15]; const float* be1 = (const float*)d_in[16];
  const float* We2 = (const float*)d_in[17]; const float* be2 = (const float*)d_in[18];

  float* credit = (float*)d_out;                 // [B,S]
  float* cm     = (float*)d_out + BS_;           // [B,S,S]

  char* ws = (char*)d_ws;
  size_t off = 0;
  auto take = [&](size_t bytes) -> void* {
    void* p = ws + off;
    off += (bytes + 255) & ~(size_t)255;
    return p;
  };
  _Float16* xh   = (_Float16*)take((size_t)BS_ * H_ * 2);   // 8 MB
  _Float16* Wqks = (_Float16*)take((size_t)H_ * 512 * 2);   // swizzled Q|K weight
  _Float16* Wm1s = (_Float16*)take((size_t)H_ * 128 * 2);
  _Float16* Wi1s = (_Float16*)take((size_t)H_ * H_ * 2);
  _Float16* Wi2s = (_Float16*)take((size_t)H_ * H_ * 2);
  _Float16* We1s = (_Float16*)take((size_t)H_ * 128 * 2);
  float* bqk  = (float*)take(512 * 4);
  float* aw   = (float*)take((size_t)B_ * S_ * 4);
  float* strg = (float*)take((size_t)B_ * S_ * 4);
  float* clm  = (float*)take((size_t)B_ * S_ * 4);
  float* base = (float*)take((size_t)B_ * H_ * 4);
  _Float16* QKh = (_Float16*)take((size_t)BS_ * 512 * 2);   // 16 MB

  // 1) precision conversions + weight swizzles (fragment-native order)
  cvt4_f16_kernel<<<(BS_ * H_ / 4 + 255) / 256, 256, 0, stream>>>(x, xh, BS_ * H_ / 4);
  swz_b_kernel<<<(256 * 256 + 255) / 256, 256, 0, stream>>>(Wq, Wqks, 256, 32, 0);
  swz_b_kernel<<<(256 * 256 + 255) / 256, 256, 0, stream>>>(Wk, Wqks, 256, 32, 16);
  swz_b_kernel<<<(256 * 128 + 255) / 256, 256, 0, stream>>>(Wm1, Wm1s, 128, 8, 0);
  swz_b_kernel<<<(256 * 256 + 255) / 256, 256, 0, stream>>>(Wi1, Wi1s, 256, 16, 0);
  swz_b_kernel<<<(256 * 256 + 255) / 256, 256, 0, stream>>>(Wi2, Wi2s, 256, 16, 0);
  swz_b_kernel<<<(256 * 128 + 255) / 256, 256, 0, stream>>>(We1, We1s, 128, 8, 0);
  concat_bias_kernel<<<2, 256, 0, stream>>>(bq, bk, bqk);
  zero_f32_kernel<<<(B_ * S_) / 256, 256, 0, stream>>>(aw, B_ * S_);

  // 2) fused Q|K projection (WMMA)
  proj_qk_kernel<<<BS_ / 16, 256, 0, stream>>>(xh, Wqks, bqk, QKh);

  // 3) strengths MLP (WMMA stage-1)
  strengths_kernel<<<BS_ / 16, 256, 0, stream>>>(xh, Wm1s, bm1, Wm2, bm2, strg);

  // 4) attention softmax column sums (WMMA, flash-style)
  attn_colsum_kernel<<<B_ * NH_ * (S_ / 16), 256, 0, stream>>>(QKh, aw);

  // 5) colmean / aw_mean, then cm streaming write and base reduction
  colmean_kernel<<<B_, 256, 0, stream>>>(aw, strg, clm);
  cm_kernel<<<B_ * S_, 256, 0, stream>>>(aw, strg, cm);
  base_kernel<<<B_, 256, 0, stream>>>(x, clm, base);

  // 6) fused intervention MLP + credit head (3 chained WMMA GEMMs in LDS)
  mlp_credit_kernel<<<BS_ / 16, 256, 0, stream>>>(
      xh, Wi1s, bi1, Wi2s, bi2, We1s, be1, We2, be2, base, clm, fo, mask, credit);
}
